// GNNpaper_75368086110729
// MI455X (gfx1250) — compile-verified, matrix operand order checked by reading, not run on hardware
//
#include <hip/hip_runtime.h>
#include <hip/hip_bf16.h>
#include <math.h>

typedef __attribute__((ext_vector_type(16))) _Float16 v16h;
typedef __attribute__((ext_vector_type(8)))  _Float16 v8h;
typedef __attribute__((ext_vector_type(8)))  float    v8f;

// ---------------- problem constants ----------------
constexpr int  BGR   = 64;           // graphs
constexpr int  NPG   = 1024;         // nodes per graph
constexpr long NT    = 65536;        // total nodes
constexpr long E_TOT = 2097152;      // total edges
constexpr long E_PG  = 32768;        // edges per graph

// ---------------- workspace layout (bytes) ----------------
constexpr size_t OFF_XH     = 0;                               // f16 [NT*128]
constexpr size_t OFF_W1L    = OFF_XH    + NT*128*2;            // f16 128*128
constexpr size_t OFF_W1R    = OFF_W1L   + 128*128*2;
constexpr size_t OFF_W2L    = OFF_W1R   + 128*128*2;           // f16 128*64
constexpr size_t OFF_W2R    = OFF_W2L   + 128*64*2;
constexpr size_t OFF_WM     = OFF_W2R   + 128*64*2;            // f16 64*64
constexpr size_t OFF_WREL   = OFF_WM    + 64*64*2;             // f16 128*128
constexpr size_t OFF_WROOT  = OFF_WREL  + 128*128*2;
constexpr size_t OFF_WC1    = OFF_WROOT + 128*128*2;           // f16 8192*128
constexpr size_t OFF_AGG    = OFF_WC1   + 8192*128*2;          // f32 [NT*128]
constexpr size_t OFF_ACC    = OFF_AGG   + NT*128*4;            // f32 [NT*128]
constexpr size_t OFF_SACC   = OFF_ACC   + NT*128*4;            // f32 [NT*64]
constexpr size_t OFF_AGGH   = OFF_SACC  + NT*64*4;             // f16 [NT*128]
constexpr size_t OFF_ZH     = OFF_AGGH  + NT*128*2;            // f16 [NT*128]
constexpr size_t OFF_S0H    = OFF_ZH    + NT*128*2;            // f16 [NT*64]
constexpr size_t OFF_SH     = OFF_S0H   + NT*64*2;             // f16 [NT*64]
constexpr size_t OFF_CNT    = OFF_SH    + NT*64*2;             // u32 [2^24] packed u8 counts
constexpr size_t OFF_DEG    = OFF_CNT   + 16777216ull*4;       // u32 [NT]
constexpr size_t OFF_ADJP   = OFF_DEG   + NT*4;                // f32 [64*64*64]
constexpr size_t OFF_ADJPH  = OFF_ADJP  + 64*64*64*4;          // f16
constexpr size_t OFF_ROWSUM = OFF_ADJPH + 64*64*64*2;          // f32 [64*64]
constexpr size_t OFF_POOL   = OFF_ROWSUM+ 64*64*4;             // f32 [64*64*128]
constexpr size_t OFF_POOLH  = OFF_POOL  + 64*64*128*4;         // f16
constexpr size_t OFF_GBUF   = OFF_POOLH + 64*64*128*2;         // f32 [64*64*64]
constexpr size_t OFF_T1     = OFF_GBUF  + 64*64*64*4;          // f32 [64*64*128]
constexpr size_t OFF_AGGPH  = OFF_T1    + 64*64*128*4;         // f16
constexpr size_t OFF_HACC   = OFF_AGGPH + 64*64*128*2;         // f32 [64*8192]
constexpr size_t OFF_HH     = OFF_HACC  + 64*8192*4;           // f16
constexpr size_t OFF_H1     = OFF_HH    + 64*8192*2;           // f32 [64*128]
constexpr size_t OFF_SCAL   = OFF_H1    + 64*128*4;            // f32 [8]: sumA2, trace, gram, ent

#define WSP(off, T) ((T*)((char*)d_ws + (off)))

// ---------------- WMMA helpers ----------------
// A/B fragment from LDS tile [rows][40] (K-contiguous per row, 16B-aligned rows).
// lane L: row uses (L&15), halves: K in [8h,8h+8) and [16+8h,16+8h+8), h=L>>4.
__device__ __forceinline__ v16h load_frag(const _Float16 tile[][40], int row, int h) {
    const _Float16* p = &tile[row][0];
    v8h lo = *(const v8h*)(p + 8*h);
    v8h hi = *(const v8h*)(p + 16 + 8*h);
    v16h r;
#pragma unroll
    for (int i = 0; i < 8; ++i) { r[i] = lo[i]; r[i+8] = hi[i]; }
    return r;
}
__device__ __forceinline__ v8f wmma_f16(v16h a, v16h b, v8f c) {
    return __builtin_amdgcn_wmma_f32_16x16x32_f16(false, a, false, b, (short)0, c, false, false);
}
__device__ __forceinline__ float leaky(float v) { return v > 0.f ? v : 0.01f*v; }

// ---------------- generic 64x64-tile WMMA GEMM ----------------
// C[M,N](f32) (=|+=) op(A)(f16) x B[K,N](f16).  M%64==0, N%64==0, K%32==0.
// TRANSA: A stored [K, M] (lda = row stride over k).
template<bool TRANSA, bool ACC>
__global__ __launch_bounds__(256) void gemm_wmma_kernel(
    const _Float16* __restrict__ A, long aBS, int lda,
    const _Float16* __restrict__ B, long bBS, int ldb,
    float* __restrict__ C, long cBS, int ldc,
    int M, int N, int K)
{
    __shared__ __align__(16) _Float16 At[64][40];
    __shared__ __align__(16) _Float16 Bt[64][40];
    const int bm = blockIdx.x * 64;
    const int bn = blockIdx.y * 64;
    A += (long)blockIdx.z * aBS;
    B += (long)blockIdx.z * bBS;
    C += (long)blockIdx.z * cBS;
    const int t = threadIdx.x, wave = t >> 5, lane = t & 31;
    const int ln = lane & 15, hh = lane >> 4;
    const int tn = wave & 3, tm0 = wave >> 2;        // wave owns tiles (tm0,tn),(tm0+2,tn)
    v8f acc0 = {}, acc1 = {};
    for (int k0 = 0; k0 < K; k0 += 32) {
        __syncthreads();
        if (!TRANSA) {
            int r = t >> 2, seg = t & 3;
            uint4 v = *(const uint4*)(A + (long)(bm + r)*lda + k0 + seg*8);
            *(uint4*)&At[r][seg*8] = v;
        } else {
            int kk = t >> 3, seg = t & 7;
            uint4 v = *(const uint4*)(A + (long)(k0 + kk)*lda + bm + seg*8);
            const _Float16* hv = (const _Float16*)&v;
#pragma unroll
            for (int i = 0; i < 8; ++i) At[seg*8 + i][kk] = hv[i];
        }
        {
            int kk = t >> 3, seg = t & 7;
            uint4 v = *(const uint4*)(B + (long)(k0 + kk)*ldb + bn + seg*8);
            const _Float16* hv = (const _Float16*)&v;
#pragma unroll
            for (int i = 0; i < 8; ++i) Bt[seg*8 + i][kk] = hv[i];
        }
        __syncthreads();
        v16h bf = load_frag(Bt, tn*16 + ln, hh);
        v16h a0 = load_frag(At, tm0*16 + ln, hh);
        v16h a1 = load_frag(At, (tm0+2)*16 + ln, hh);
        acc0 = wmma_f16(a0, bf, acc0);
        acc1 = wmma_f16(a1, bf, acc1);
    }
#pragma unroll
    for (int v = 0; v < 8; ++v) {
        int m0 = bm + tm0*16 + v + 8*hh;
        int m1 = bm + (tm0+2)*16 + v + 8*hh;
        int n  = bn + tn*16 + ln;
        long i0 = (long)m0*ldc + n, i1 = (long)m1*ldc + n;
        if (ACC) { C[i0] += acc0[v]; C[i1] += acc1[v]; }
        else     { C[i0]  = acc0[v]; C[i1]  = acc1[v]; }
    }
}

// ---------------- adj_p = sum_edges s_src (x) s_dst, per graph (WMMA over gathered rows) ----
__global__ __launch_bounds__(256) void adjp_edge_kernel(
    const int* __restrict__ esrc, const int* __restrict__ edst,
    const _Float16* __restrict__ S,         // [NT,64] softmaxed s (f16)
    float* __restrict__ adjp,               // [64,64,64]
    float* __restrict__ trace_accum)        // SCAL+1
{
    __shared__ __align__(16) _Float16 At[64][40];   // At[k][e] = s[src_e][k]
    __shared__ __align__(16) _Float16 Bt[64][40];   // Bt[l][e] = s[dst_e][l]
    const int g = blockIdx.x;
    const int* src = esrc + (long)g * E_PG;
    const int* dst = edst + (long)g * E_PG;
    const int t = threadIdx.x, wave = t >> 5, lane = t & 31;
    const int ln = lane & 15, hh = lane >> 4;
    const int tn = wave & 3, tm0 = wave >> 2;
    const int e = t >> 3, seg = t & 7;
    v8f acc0 = {}, acc1 = {};
    for (long c0 = 0; c0 < E_PG; c0 += 32) {
        __syncthreads();
        int si = src[c0 + e];
        int di = dst[c0 + e];
        uint4 vs = *(const uint4*)(S + (long)si*64 + seg*8);
        uint4 vd = *(const uint4*)(S + (long)di*64 + seg*8);
        const _Float16* hs = (const _Float16*)&vs;
        const _Float16* hd = (const _Float16*)&vd;
#pragma unroll
        for (int i = 0; i < 8; ++i) { At[seg*8 + i][e] = hs[i]; Bt[seg*8 + i][e] = hd[i]; }
        __syncthreads();
        v16h bf = load_frag(Bt, tn*16 + ln, hh);
        v16h a0 = load_frag(At, tm0*16 + ln, hh);
        v16h a1 = load_frag(At, (tm0+2)*16 + ln, hh);
        acc0 = wmma_f16(a0, bf, acc0);
        acc1 = wmma_f16(a1, bf, acc1);
    }
    float tr = 0.f;
#pragma unroll
    for (int v = 0; v < 8; ++v) {
        int m0 = tm0*16 + v + 8*hh;
        int m1 = (tm0+2)*16 + v + 8*hh;
        int n  = tn*16 + ln;
        adjp[(long)g*4096 + m0*64 + n] = acc0[v]; if (m0 == n) tr += acc0[v];
        adjp[(long)g*4096 + m1*64 + n] = acc1[v]; if (m1 == n) tr += acc1[v];
    }
    if (tr != 0.f) atomicAdd(trace_accum, tr);
}

// ---------------- elementwise / sparse / reduction kernels ----------------
__global__ void zero_u32_kernel(unsigned* p, size_t n) {
    size_t i = (size_t)blockIdx.x*blockDim.x + threadIdx.x;
    size_t st = (size_t)gridDim.x*blockDim.x;
    for (; i < n; i += st) p[i] = 0u;
}

__global__ void cast_f32_f16_kernel(const float* __restrict__ in, _Float16* __restrict__ out, size_t n) {
    size_t i = (size_t)blockIdx.x*blockDim.x + threadIdx.x;
    size_t st = (size_t)gridDim.x*blockDim.x;
    for (; i < n; i += st) out[i] = (_Float16)in[i];
}

// per edge: deg[dst]++, packed byte count for duplicate detection
__global__ void edge_count_kernel(const int* __restrict__ src, const int* __restrict__ dst,
                                  unsigned* __restrict__ cnt, unsigned* __restrict__ deg) {
    long e = (long)blockIdx.x*blockDim.x + threadIdx.x;
    if (e >= E_TOT) return;
    int si = src[e], di = dst[e];
    atomicAdd(&deg[di], 1u);
    int g = si >> 10;
    long key = ((long)g << 20) | ((long)(si & 1023) << 10) | (long)(di & 1023);
    atomicAdd(&cnt[key >> 2], 1u << ((key & 3) * 8));
}

// sum over edges of count at own slot == sum of count^2
__global__ void count_sq_kernel(const int* __restrict__ src, const int* __restrict__ dst,
                                const unsigned* __restrict__ cnt, float* __restrict__ accum) {
    long e = (long)blockIdx.x*blockDim.x + threadIdx.x;
    float c = 0.f;
    if (e < E_TOT) {
        int si = src[e], di = dst[e];
        int g = si >> 10;
        long key = ((long)g << 20) | ((long)(si & 1023) << 10) | (long)(di & 1023);
        c = (float)((cnt[key >> 2] >> ((key & 3) * 8)) & 255u);
    }
#pragma unroll
    for (int off = 16; off; off >>= 1) c += __shfl_xor(c, off, 32);
    if ((threadIdx.x & 31) == 0 && c != 0.f) atomicAdd(accum, c);
}

// scatter features: agg[dst] += feat[src]; feat is f32 (HALF=false) or f16 (HALF=true), 128 wide
template<bool HALF>
__global__ __launch_bounds__(256) void scatter_kernel(const int* __restrict__ src, const int* __restrict__ dst,
                                                      const void* __restrict__ feat, float* __restrict__ agg) {
    long tid = (long)blockIdx.x*blockDim.x + threadIdx.x;
    long e = tid >> 5; int lane = tid & 31;
    if (e >= E_TOT) return;
    int si = src[e], di = dst[e];
    float* o = agg + (long)di*128 + lane*4;
    if (HALF) {
        const _Float16* x = (const _Float16*)feat + (long)si*128 + lane*4;
#pragma unroll
        for (int i = 0; i < 4; ++i) atomicAdd(&o[i], (float)x[i]);
    } else {
        float4 v = *((const float4*)((const float*)feat + (long)si*128) + lane);
        atomicAdd(&o[0], v.x); atomicAdd(&o[1], v.y);
        atomicAdd(&o[2], v.z); atomicAdd(&o[3], v.w);
    }
}

// agg_h[n,f] = agg[n,f] / max(deg[n],1)   (128 wide)
__global__ void agg_norm_cast_kernel(const float* __restrict__ agg, const unsigned* __restrict__ deg,
                                     _Float16* __restrict__ out, size_t n) {
    size_t i = (size_t)blockIdx.x*blockDim.x + threadIdx.x;
    size_t st = (size_t)gridDim.x*blockDim.x;
    for (; i < n; i += st) {
        unsigned d = deg[i >> 7];
        out[i] = (_Float16)(agg[i] / (float)(d > 0 ? d : 1u));
    }
}

// out_h[i] = leaky(acc[i] + bias[i & mask])
__global__ void bias_leaky_cast_kernel(const float* __restrict__ acc, const float* __restrict__ bias,
                                       _Float16* __restrict__ out, int mask, size_t n) {
    size_t i = (size_t)blockIdx.x*blockDim.x + threadIdx.x;
    size_t st = (size_t)gridDim.x*blockDim.x;
    for (; i < n; i += st) out[i] = (_Float16)leaky(acc[i] + bias[i & mask]);
}

// row softmax over 64 cols of (acc + bm); writes f16 probs + accumulates entropy sum
__global__ __launch_bounds__(256) void softmax_ent_kernel(const float* __restrict__ acc, const float* __restrict__ bm,
                                                          _Float16* __restrict__ sh, float* __restrict__ ent_accum) {
    long row = (long)blockIdx.x * 8 + (threadIdx.x >> 5);
    int lane = threadIdx.x & 31;
    const float* r = acc + row*64;
    float v0 = r[lane] + bm[lane];
    float v1 = r[lane + 32] + bm[lane + 32];
    float m = fmaxf(v0, v1);
#pragma unroll
    for (int off = 16; off; off >>= 1) m = fmaxf(m, __shfl_xor(m, off, 32));
    float e0 = __expf(v0 - m), e1 = __expf(v1 - m);
    float s = e0 + e1;
#pragma unroll
    for (int off = 16; off; off >>= 1) s += __shfl_xor(s, off, 32);
    float inv = 1.f / s;
    float p0 = e0*inv, p1 = e1*inv;
    sh[row*64 + lane]      = (_Float16)p0;
    sh[row*64 + lane + 32] = (_Float16)p1;
    float ent = -p0*__logf(p0 + 1e-15f) - p1*__logf(p1 + 1e-15f);
#pragma unroll
    for (int off = 16; off; off >>= 1) ent += __shfl_xor(ent, off, 32);
    if (lane == 0) atomicAdd(ent_accum, ent);
}

// sum of squares reduction (for ||s^T s||_F^2)
__global__ void sq_reduce_kernel(const float* __restrict__ in, size_t n, float* __restrict__ accum) {
    size_t i = (size_t)blockIdx.x*blockDim.x + threadIdx.x;
    size_t st = (size_t)gridDim.x*blockDim.x;
    float s = 0.f;
    for (; i < n; i += st) { float v = in[i]; s += v*v; }
#pragma unroll
    for (int off = 16; off; off >>= 1) s += __shfl_xor(s, off, 32);
    if ((threadIdx.x & 31) == 0) atomicAdd(accum, s);
}

// adj_p rowsum + f16 cast; grid=64 graphs, block=64 (one row each)
__global__ void adjp_post_kernel(const float* __restrict__ adjp, _Float16* __restrict__ adjph,
                                 float* __restrict__ rowsum) {
    int g = blockIdx.x, k = threadIdx.x;
    const float* r = adjp + (long)g*4096 + k*64;
    _Float16* o = adjph + (long)g*4096 + k*64;
    float s = 0.f;
#pragma unroll 8
    for (int j = 0; j < 64; ++j) { float v = r[j]; s += v; o[j] = (_Float16)v; }
    rowsum[g*64 + k] = s;
}

// aggp_h[g,k,f] = t1[g,k,f] / max(rowsum[g,k],1)
__global__ void aggp_div_kernel(const float* __restrict__ t1, const float* __restrict__ rowsum,
                                _Float16* __restrict__ out, size_t n) {
    size_t i = (size_t)blockIdx.x*blockDim.x + threadIdx.x;
    size_t st = (size_t)gridDim.x*blockDim.x;
    for (; i < n; i += st) {
        int g = (int)(i >> 13), k = (int)((i >> 7) & 63);
        out[i] = (_Float16)(t1[i] / fmaxf(rowsum[g*64 + k], 1.f));
    }
}

// final classifier: logits[g] = leaky(H1[g,:]+bc1) . Wc2 + bc2
__global__ void classify_kernel(const float* __restrict__ h1, const float* __restrict__ bc1,
                                const float* __restrict__ wc2, const float* __restrict__ bc2,
                                float* __restrict__ out) {
    __shared__ float red[128];
    int g = blockIdx.x, t = threadIdx.x;
    float v = leaky(h1[g*128 + t] + bc1[t]);
    red[t] = v * wc2[t];
    __syncthreads();
    for (int s = 64; s; s >>= 1) { if (t < s) red[t] += red[t + s]; __syncthreads(); }
    if (t == 0) out[g] = red[0] + bc2[0];
}

__global__ void loss_kernel(const float* __restrict__ scal, float* __restrict__ out) {
    float v = scal[0] - 2.f*scal[1] + scal[2];
    float link = (v > 0.f ? sqrtf(v) : 0.f) / 67108864.f;  // a.size = 64*1024*1024
    out[0] = link + scal[3] / 65536.f;                     // + mean entropy
}

// ---------------- host-side orchestration ----------------
static inline void launch_gemm(bool transA, bool acc,
                               const _Float16* A, long aBS, int lda,
                               const _Float16* B, long bBS, int ldb,
                               float* C, long cBS, int ldc,
                               int M, int N, int K, int batch, hipStream_t s) {
    dim3 grid(M/64, N/64, batch);
    if (!transA && !acc) gemm_wmma_kernel<false,false><<<grid,256,0,s>>>(A,aBS,lda,B,bBS,ldb,C,cBS,ldc,M,N,K);
    else if (!transA && acc) gemm_wmma_kernel<false,true ><<<grid,256,0,s>>>(A,aBS,lda,B,bBS,ldb,C,cBS,ldc,M,N,K);
    else if ( transA && !acc) gemm_wmma_kernel<true ,false><<<grid,256,0,s>>>(A,aBS,lda,B,bBS,ldb,C,cBS,ldc,M,N,K);
}

extern "C" void kernel_launch(void* const* d_in, const int* in_sizes, int n_in,
                              void* d_out, int out_size, void* d_ws, size_t ws_size,
                              hipStream_t stream) {
    const float* x     = (const float*)d_in[0];
    const int*   eidx  = (const int*)d_in[1];
    const int*   esrc  = eidx;
    const int*   edst  = eidx + E_TOT;
    const float* W1l   = (const float*)d_in[5];
    const float* b1l   = (const float*)d_in[6];
    const float* W1r   = (const float*)d_in[7];
    const float* W2l   = (const float*)d_in[8];
    const float* b2l   = (const float*)d_in[9];
    const float* W2r   = (const float*)d_in[10];
    const float* Wm    = (const float*)d_in[11];
    const float* bm    = (const float*)d_in[12];
    const float* Wrel  = (const float*)d_in[13];
    const float* Wroot = (const float*)d_in[14];
    const float* broot = (const float*)d_in[15];
    const float* Wc1   = (const float*)d_in[16];
    const float* bc1   = (const float*)d_in[17];
    const float* Wc2   = (const float*)d_in[18];
    const float* bc2   = (const float*)d_in[19];
    float* out = (float*)d_out;

    _Float16* XH    = WSP(OFF_XH, _Float16);
    _Float16* W1LH  = WSP(OFF_W1L, _Float16);
    _Float16* W1RH  = WSP(OFF_W1R, _Float16);
    _Float16* W2LH  = WSP(OFF_W2L, _Float16);
    _Float16* W2RH  = WSP(OFF_W2R, _Float16);
    _Float16* WMH   = WSP(OFF_WM, _Float16);
    _Float16* WRELH = WSP(OFF_WREL, _Float16);
    _Float16* WROOTH= WSP(OFF_WROOT, _Float16);
    _Float16* WC1H  = WSP(OFF_WC1, _Float16);
    float*    AGG   = WSP(OFF_AGG, float);
    float*    ACC   = WSP(OFF_ACC, float);
    float*    SACC  = WSP(OFF_SACC, float);
    _Float16* AGGH  = WSP(OFF_AGGH, _Float16);
    _Float16* ZH    = WSP(OFF_ZH, _Float16);
    _Float16* S0H   = WSP(OFF_S0H, _Float16);
    _Float16* SH    = WSP(OFF_SH, _Float16);
    unsigned* CNT   = WSP(OFF_CNT, unsigned);
    unsigned* DEG   = WSP(OFF_DEG, unsigned);
    float*    ADJP  = WSP(OFF_ADJP, float);
    _Float16* ADJPH = WSP(OFF_ADJPH, _Float16);
    float*    ROWSUM= WSP(OFF_ROWSUM, float);
    float*    POOL  = WSP(OFF_POOL, float);
    _Float16* POOLH = WSP(OFF_POOLH, _Float16);
    float*    GBUF  = WSP(OFF_GBUF, float);
    float*    T1    = WSP(OFF_T1, float);
    _Float16* AGGPH = WSP(OFF_AGGPH, _Float16);
    float*    HACC  = WSP(OFF_HACC, float);
    _Float16* HH    = WSP(OFF_HH, _Float16);
    float*    H1    = WSP(OFF_H1, float);
    float*    SCAL  = WSP(OFF_SCAL, float);

    // 0) zero accumulators
    zero_u32_kernel<<<4096,256,0,stream>>>((unsigned*)AGG, NT*128);
    zero_u32_kernel<<<256,256,0,stream>>>(DEG, NT);
    zero_u32_kernel<<<4096,256,0,stream>>>(CNT, 16777216ull);
    zero_u32_kernel<<<1,32,0,stream>>>((unsigned*)SCAL, 8);

    // 1) f16 casts of x and all weights
    cast_f32_f16_kernel<<<4096,256,0,stream>>>(x, XH, NT*128);
    cast_f32_f16_kernel<<<64,256,0,stream>>>(W1l, W1LH, 128*128);
    cast_f32_f16_kernel<<<64,256,0,stream>>>(W1r, W1RH, 128*128);
    cast_f32_f16_kernel<<<32,256,0,stream>>>(W2l, W2LH, 128*64);
    cast_f32_f16_kernel<<<32,256,0,stream>>>(W2r, W2RH, 128*64);
    cast_f32_f16_kernel<<<16,256,0,stream>>>(Wm, WMH, 64*64);
    cast_f32_f16_kernel<<<64,256,0,stream>>>(Wrel, WRELH, 128*128);
    cast_f32_f16_kernel<<<64,256,0,stream>>>(Wroot, WROOTH, 128*128);
    cast_f32_f16_kernel<<<4096,256,0,stream>>>(Wc1, WC1H, 8192*128);

    // 2) edge pass: degrees + packed duplicate counts; sum c^2; scatter x
    edge_count_kernel<<<(int)(E_TOT/256),256,0,stream>>>(esrc, edst, CNT, DEG);
    scatter_kernel<false><<<(int)(E_TOT*32/256),256,0,stream>>>(esrc, edst, (const void*)x, AGG);
    count_sq_kernel<<<(int)(E_TOT/256),256,0,stream>>>(esrc, edst, CNT, &SCAL[0]);

    // 3) SAGE layer 1: z = leaky(mean_agg @ W1l + x @ W1r + b1l)
    agg_norm_cast_kernel<<<8192,256,0,stream>>>(AGG, DEG, AGGH, NT*128);
    launch_gemm(false,false, AGGH,0,128, W1LH,0,128, ACC,0,128, (int)NT,128,128, 1, stream);
    launch_gemm(false,true,  XH,  0,128, W1RH,0,128, ACC,0,128, (int)NT,128,128, 1, stream);
    bias_leaky_cast_kernel<<<8192,256,0,stream>>>(ACC, b1l, ZH, 127, NT*128);

    // 4) SAGE layer 2 + Wm: s_pre = leaky(mean_agg(z) @ W2l + z @ W2r + b2l) @ Wm + bm
    zero_u32_kernel<<<4096,256,0,stream>>>((unsigned*)AGG, NT*128);
    scatter_kernel<true><<<(int)(E_TOT*32/256),256,0,stream>>>(esrc, edst, (const void*)ZH, AGG);
    agg_norm_cast_kernel<<<8192,256,0,stream>>>(AGG, DEG, AGGH, NT*128);
    launch_gemm(false,false, AGGH,0,128, W2LH,0,64, SACC,0,64, (int)NT,64,128, 1, stream);
    launch_gemm(false,true,  ZH,  0,128, W2RH,0,64, SACC,0,64, (int)NT,64,128, 1, stream);
    bias_leaky_cast_kernel<<<8192,256,0,stream>>>(SACC, b2l, S0H, 63, NT*64);
    launch_gemm(false,false, S0H,0,64, WMH,0,64, ACC,0,64, (int)NT,64,64, 1, stream);

    // 5) softmax + entropy -> s (f16)
    softmax_ent_kernel<<<8192,256,0,stream>>>(ACC, bm, SH, &SCAL[3]);

    // 6) pooled = s^T z  [B,64,128];  G = s^T s  [B,64,64] -> ||G||^2
    launch_gemm(true,false, SH,(long)NPG*64,64, ZH,(long)NPG*128,128, POOL,64*128,128, 64,128,NPG, BGR, stream);
    launch_gemm(true,false, SH,(long)NPG*64,64, SH,(long)NPG*64,64,   GBUF,64*64,64,   64,64, NPG, BGR, stream);
    sq_reduce_kernel<<<1024,256,0,stream>>>(GBUF, 64ull*4096, &SCAL[2]);

    // 7) adj_p via edge outer-product WMMA; trace -> SCAL[1]
    adjp_edge_kernel<<<BGR,256,0,stream>>>(esrc, edst, SH, ADJP, &SCAL[1]);
    adjp_post_kernel<<<BGR,64,0,stream>>>(ADJP, ADJPH, ROWSUM);
    cast_f32_f16_kernel<<<2048,256,0,stream>>>(POOL, POOLH, 64ull*64*128);

    // 8) DenseSAGEConv: h = leaky((adj_p@pooled/rowsum)@Wrel + pooled@Wroot + broot)
    launch_gemm(false,false, ADJPH,4096,64, POOLH,64*128,128, T1,64*128,128, 64,128,64, BGR, stream);
    aggp_div_kernel<<<2048,256,0,stream>>>(T1, ROWSUM, AGGPH, 64ull*64*128);
    launch_gemm(false,false, AGGPH,64*128,128, WRELH,0,128,  HACC,64*128,128, 64,128,128, BGR, stream);
    launch_gemm(false,true,  POOLH,64*128,128, WROOTH,0,128, HACC,64*128,128, 64,128,128, BGR, stream);
    bias_leaky_cast_kernel<<<2048,256,0,stream>>>(HACC, broot, HH, 127, 64ull*8192);

    // 9) classifier: logits = leaky(h @ Wc1 + bc1) @ Wc2 + bc2
    launch_gemm(false,false, HH,0,8192, WC1H,0,128, H1,0,128, 64,128,8192, 1, stream);
    classify_kernel<<<BGR,128,0,stream>>>(H1, bc1, Wc2, bc2, out);

    // 10) loss = sqrt(sumA2 - 2*trace + gram)/a.size + ent/NT
    loss_kernel<<<1,1,0,stream>>>(SCAL, out + 64);
}
